// SwinTransformerBlock_15247133900943
// MI455X (gfx1250) — matrix operand order repaired and checked by reading, not compiled, not run on hardware
//
#include <hip/hip_runtime.h>
#include <hip/hip_bf16.h>
#include <math.h>

// ---------------------------------------------------------------------------
// Types for WMMA fragments (gfx1250, wave32)
// ---------------------------------------------------------------------------
typedef __attribute__((ext_vector_type(16))) __bf16 bf16x16;
typedef __attribute__((ext_vector_type(8)))  float  f32x8;
typedef __attribute__((ext_vector_type(4))) unsigned uint32x4;
typedef __attribute__((ext_vector_type(8))) int      int32x8;
typedef __attribute__((ext_vector_type(4))) int      int32x4;
typedef unsigned short u16;

struct U4 { unsigned x, y, z, w; };          // 16-byte chunk (8 bf16)
struct Frag { U4 lo, hi; };                  // 32 bytes = 16 bf16

static __device__ __forceinline__ bf16x16 make_frag(U4 lo, U4 hi) {
    Frag f; f.lo = lo; f.hi = hi;
    return __builtin_bit_cast(bf16x16, f);
}

static __device__ __forceinline__ u16 f2bf(float f) {
    unsigned u = __builtin_bit_cast(unsigned, f);
    u += 0x7FFFu + ((u >> 16) & 1u);         // round-to-nearest-even
    return (u16)(u >> 16);
}
static __device__ __forceinline__ float bf2f(u16 h) {
    unsigned u = ((unsigned)h) << 16;
    return __builtin_bit_cast(float, u);
}

// Low 32 bits of a flat LDS pointer == LDS byte address (ISA 10.2 aperture map)
static __device__ __forceinline__ unsigned lds_addr_of(const void* p) {
    return (unsigned)(unsigned long long)p;
}

// ---------------------------------------------------------------------------
// Problem constants
// ---------------------------------------------------------------------------
#define BB   16
#define HH   56
#define CC   384
#define NHd  12
#define HDd  32
#define NN   64          // tokens per window (8x8)
#define NWIN 49          // windows per image (7x7)
#define ROWS 50176       // B * H * W
#define HID  1536

// ---------------------------------------------------------------------------
// TDM: issue a 2-D tile load (tile_dim0=32 elems contiguous, tile_dim1=128
// rows, row stride = K elements, bf16) into LDS.  Wave-level op, EXEC ignored.
// ---------------------------------------------------------------------------
static __device__ __forceinline__ int32x8 tdm_group1(int M, int K) {
    int32x8 g1;
    unsigned td0 = (unsigned)K;                // tensor dim0 (elements)
    unsigned td1 = (unsigned)M;                // tensor dim1 (rows)
    unsigned long long st0 = (unsigned long long)(unsigned)K;  // dim0 stride
    g1[0] = (int)(1u << 16);                                   // data_size=2B
    g1[1] = (int)((td0 & 0xFFFFu) << 16);                      // tdim0 lo
    g1[2] = (int)((td0 >> 16) | ((td1 & 0xFFFFu) << 16));      // tdim0 hi | tdim1 lo
    g1[3] = (int)((td1 >> 16) | (32u << 16));                  // tdim1 hi | tile0=32
    g1[4] = (int)(128u);                                       // tile1=128, tile2=0
    g1[5] = (int)(unsigned)(st0 & 0xFFFFFFFFull);              // stride0 lo
    g1[6] = (int)(unsigned)((st0 >> 32) & 0xFFFFull);          // stride0 hi | stride1 lo
    g1[7] = 0;                                                 // stride1 hi
    return g1;
}

static __device__ __forceinline__ void tdm_load(const u16* gptr, unsigned ldsoff,
                                                int32x8 g1) {
    unsigned long long ga = (unsigned long long)gptr;
    uint32x4 g0;
    g0[0] = 1u;                                                // count=1
    g0[1] = ldsoff;                                            // lds_addr
    g0[2] = (unsigned)(ga & 0xFFFFFFFFull);                    // global lo
    g0[3] = (unsigned)((ga >> 32) & 0x1FFFFFFull) | (2u << 30);// global hi | type=2
    int32x4 z = {0, 0, 0, 0};
#if __clang_major__ >= 23
    int32x8 z8 = {0, 0, 0, 0, 0, 0, 0, 0};
    __builtin_amdgcn_tensor_load_to_lds(g0, g1, z, z, z8, 0);
#else
    __builtin_amdgcn_tensor_load_to_lds(g0, g1, z, z, 0);
#endif
}

// ---------------------------------------------------------------------------
// LayerNorm (fp32 in, bf16 out).  windowed=1: fused shift-roll + window
// partition (out row is the windowed row index, input gathered).
// ---------------------------------------------------------------------------
__global__ __launch_bounds__(128)
void ln_kernel(const float* __restrict__ xin, const float* __restrict__ gam,
               const float* __restrict__ bet, u16* __restrict__ out, int windowed)
{
    int orow = blockIdx.x;
    int irow;
    if (windowed) {
        int b   = orow / (NWIN * NN);
        int win = (orow >> 6) % NWIN;
        int idx = orow & 63;
        int hs  = (win / 7) * 8 + (idx >> 3);
        int wsh = (win % 7) * 8 + (idx & 7);
        int h = hs + 4;  if (h >= HH) h -= HH;   // inverse of roll(-SHIFT)
        int w = wsh + 4; if (w >= HH) w -= HH;
        irow = b * (HH * HH) + h * HH + w;
    } else {
        irow = orow;
    }
    const float* px = xin + (size_t)irow * CC;
    int tid = threadIdx.x;
    float vals[3];
    float s = 0.f;
    #pragma unroll
    for (int i = 0; i < 3; ++i) { vals[i] = px[tid + i * 128]; s += vals[i]; }

    __shared__ float red[128];
    red[tid] = s; __syncthreads();
    for (int st = 64; st > 0; st >>= 1) { if (tid < st) red[tid] += red[tid + st]; __syncthreads(); }
    float mean = red[0] * (1.0f / CC);
    __syncthreads();
    s = 0.f;
    #pragma unroll
    for (int i = 0; i < 3; ++i) { float d = vals[i] - mean; s += d * d; }
    red[tid] = s; __syncthreads();
    for (int st = 64; st > 0; st >>= 1) { if (tid < st) red[tid] += red[tid + st]; __syncthreads(); }
    float inv = rsqrtf(red[0] * (1.0f / CC) + 1e-5f);

    u16* po = out + (size_t)orow * CC;
    #pragma unroll
    for (int i = 0; i < 3; ++i) {
        int c = tid + i * 128;
        po[c] = f2bf((vals[i] - mean) * inv * gam[c] + bet[c]);
    }
}

// ---------------------------------------------------------------------------
// fp32 -> bf16 weight conversion WITH transpose: in [K][N] -> out [N][K]
// ---------------------------------------------------------------------------
__global__ __launch_bounds__(256)
void convert_transpose_kernel(const float* __restrict__ in, u16* __restrict__ out,
                              int K, int N)
{
    int i = blockIdx.x * 256 + threadIdx.x;
    if (i < K * N) {
        int k = i / N, n = i % N;
        out[(size_t)n * K + k] = f2bf(in[i]);
    }
}

// ---------------------------------------------------------------------------
// Tiled bf16 WMMA GEMM: C[M,N] = A[M,K] * Wt[N,K]^T + bias, fused epilogues.
// TDM feeds both 128x32 tiles into double-buffered LDS; wave 0 issues DMAs,
// s_wait_tensorcnt gates the pipeline; all 8 waves run pure WMMA.
//   MODE 0: out bf16                       (QKV)
//   MODE 1: window-reverse + roll scatter, y = x + val  (proj)   fp32 out
//   MODE 2: GELU(val) -> bf16              (fc1)
//   MODE 3: val + y residual -> fp32 d_out (fc2)
// Block: 256 threads = 8 waves; 128x128 tile; wave tile 64x32 (4x2 subtiles).
// ---------------------------------------------------------------------------
template<int MODE>
__global__ __launch_bounds__(256)
void gemm_wmma(const u16* __restrict__ A, const u16* __restrict__ Wt,
               const float* __restrict__ bias, int M, int N, int K,
               u16* __restrict__ outH, float* __restrict__ outF,
               const float* __restrict__ res1, const float* __restrict__ res2)
{
    __shared__ u16 lA[2][128 * 32];   // A tile  [row][k]
    __shared__ u16 lB[2][128 * 32];   // Wt tile [n][k]  (B already transposed)

    const int tid   = threadIdx.x;
    const int lane  = tid & 31;
    const int wave  = tid >> 5;
    const int hlf   = lane >> 4;      // lane half (VGPR layout select)
    const int ln16  = lane & 15;
    const int waveM = wave >> 2;      // 0..1  (64 rows each)
    const int waveN = wave & 3;       // 0..3  (32 cols each)
    const int tileM = blockIdx.y * 128;
    const int tileN = blockIdx.x * 128;
    const bool issuer = (wave == 0);

    const int32x8 g1 = tdm_group1(M, K);
    const u16* aBase = A  + (size_t)tileM * K;
    const u16* bBase = Wt + (size_t)tileN * K;
    unsigned offA[2] = { lds_addr_of(&lA[0][0]), lds_addr_of(&lA[1][0]) };
    unsigned offB[2] = { lds_addr_of(&lB[0][0]), lds_addr_of(&lB[1][0]) };

    f32x8 zero = {};
    f32x8 acc[4][2];
    #pragma unroll
    for (int i = 0; i < 4; ++i)
        #pragma unroll
        for (int j = 0; j < 2; ++j) acc[i][j] = zero;

    const int KT = K >> 5;            // K/32 steps
    if (issuer) {
        tdm_load(aBase, offA[0], g1);
        tdm_load(bBase, offB[0], g1);
    }

    for (int kt = 0; kt < KT; ++kt) {
        const int p = kt & 1;
        if (issuer) {
            if (kt + 1 < KT) {
                tdm_load(aBase + (kt + 1) * 32, offA[1 - p], g1);
                tdm_load(bBase + (kt + 1) * 32, offB[1 - p], g1);
                __builtin_amdgcn_s_wait_tensorcnt((short)2);   // current pair done
            } else {
                __builtin_amdgcn_s_wait_tensorcnt((short)0);
            }
        }
        __syncthreads();              // tile p ready for all waves

        bf16x16 af[4];
        #pragma unroll
        for (int ms = 0; ms < 4; ++ms) {
            int row = waveM * 64 + ms * 16 + ln16;
            U4 lo = *(const U4*)(&lA[p][row * 32 + hlf * 8]);
            U4 hi = *(const U4*)(&lA[p][row * 32 + 16 + hlf * 8]);
            af[ms] = make_frag(lo, hi);
        }
        #pragma unroll
        for (int ns = 0; ns < 2; ++ns) {
            int cn = waveN * 32 + ns * 16 + ln16;
            U4 lo = *(const U4*)(&lB[p][cn * 32 + hlf * 8]);
            U4 hi = *(const U4*)(&lB[p][cn * 32 + 16 + hlf * 8]);
            bf16x16 bf = make_frag(lo, hi);
            #pragma unroll
            for (int ms = 0; ms < 4; ++ms) {
                acc[ms][ns] = __builtin_amdgcn_wmma_f32_16x16x32_bf16(
                    false, af[ms], false, bf, (short)0, acc[ms][ns], false, false);
            }
        }
        __syncthreads();              // tile p free before DMA of kt+2 reuses it
    }

    // ---- epilogue
    #pragma unroll
    for (int ms = 0; ms < 4; ++ms) {
        #pragma unroll
        for (int ns = 0; ns < 2; ++ns) {
            int col = tileN + waveN * 32 + ns * 16 + ln16;
            float bv = bias[col];
            #pragma unroll
            for (int r = 0; r < 8; ++r) {
                int row = tileM + waveM * 64 + ms * 16 + hlf * 8 + r;
                float v = acc[ms][ns][r] + bv;
                if (MODE == 0) {
                    outH[(size_t)row * N + col] = f2bf(v);
                } else if (MODE == 2) {
                    float g = 0.5f * v * (1.0f + erff(v * 0.7071067811865475f));
                    outH[(size_t)row * N + col] = f2bf(g);
                } else if (MODE == 1) {
                    // window reverse + roll(+SHIFT) scatter, residual with x
                    int idx = row & 63;
                    int win = (row >> 6) % NWIN;
                    int b   = row / (NWIN * NN);
                    int hs  = (win / 7) * 8 + (idx >> 3);
                    int wsh = (win % 7) * 8 + (idx & 7);
                    int h = hs + 4;  if (h >= HH) h -= HH;
                    int w = wsh + 4; if (w >= HH) w -= HH;
                    size_t g = ((size_t)b * (HH * HH) + h * HH + w) * CC + col;
                    outF[g] = res1[g] + v;
                } else { // MODE 3
                    size_t g = (size_t)row * CC + col;
                    outF[g] = v + res2[g];
                }
            }
        }
    }
}

// ---------------------------------------------------------------------------
// Attention: one block (128 thr = 4 waves) per (window, head).
// S = (q*scale) @ k^T + rel_bias + mask ; softmax (fp32, also written to
// d_out attn region) ; O = P @ V -> bf16 for the proj GEMM.
// ---------------------------------------------------------------------------
__global__ __launch_bounds__(128)
void attn_kernel(const u16* __restrict__ qkv, const float* __restrict__ relb,
                 const int* __restrict__ relidx, const float* __restrict__ mask,
                 float* __restrict__ attn_out, u16* __restrict__ obf)
{
    __shared__ u16  qs[NN * HDd];     // q, scaled
    __shared__ u16  ks[NN * HDd];     // k row-major (== (k^T) col-contiguous)
    __shared__ u16  vT[HDd * NN];     // v transposed [c][i]
    __shared__ float Sm[NN * NN];     // scores fp32
    __shared__ u16  Pm[NN * NN];      // softmax probs bf16

    const int blk  = blockIdx.x;
    const int win  = blk / NHd;
    const int head = blk % NHd;
    const int wr   = win % NWIN;
    const int tid  = threadIdx.x;
    const int lane = tid & 31;
    const int wv   = tid >> 5;
    const int hlf  = lane >> 4;
    const int ln16 = lane & 15;
    const float scale = 0.17677669529663687f;   // 32^-0.5

    // ---- stage q/k/v
    {
        int i  = tid >> 1;
        int c0 = (tid & 1) * 16;
        size_t base = (size_t)(win * NN + i) * (3 * CC) + head * HDd + c0;
        u16 tmp[16];
        *(U4*)(tmp)     = *(const U4*)(qkv + base);
        *(U4*)(tmp + 8) = *(const U4*)(qkv + base + 8);
        #pragma unroll
        for (int j = 0; j < 16; ++j) qs[i * HDd + c0 + j] = f2bf(bf2f(tmp[j]) * scale);
        *(U4*)(&ks[i * HDd + c0])     = *(const U4*)(qkv + base + CC);
        *(U4*)(&ks[i * HDd + c0 + 8]) = *(const U4*)(qkv + base + CC + 8);
        *(U4*)(tmp)     = *(const U4*)(qkv + base + 2 * CC);
        *(U4*)(tmp + 8) = *(const U4*)(qkv + base + 2 * CC + 8);
        #pragma unroll
        for (int j = 0; j < 16; ++j) vT[(c0 + j) * NN + i] = tmp[j];
    }
    __syncthreads();

    f32x8 zero = {};
    // ---- S = q @ k^T ; wave wv owns rows wv*16..+15
    {
        U4 alo = *(const U4*)(&qs[(wv * 16 + ln16) * HDd + hlf * 8]);
        U4 ahi = *(const U4*)(&qs[(wv * 16 + ln16) * HDd + 16 + hlf * 8]);
        bf16x16 aq = make_frag(alo, ahi);
        f32x8 sacc[4];
        #pragma unroll
        for (int ns = 0; ns < 4; ++ns) {
            U4 blo = *(const U4*)(&ks[(ns * 16 + ln16) * HDd + hlf * 8]);
            U4 bhi = *(const U4*)(&ks[(ns * 16 + ln16) * HDd + 16 + hlf * 8]);
            bf16x16 bk = make_frag(blo, bhi);
            sacc[ns] = __builtin_amdgcn_wmma_f32_16x16x32_bf16(
                false, aq, false, bk, (short)0, zero, false, false);
        }
        #pragma unroll
        for (int ns = 0; ns < 4; ++ns)
            #pragma unroll
            for (int r = 0; r < 8; ++r) {
                int i = wv * 16 + hlf * 8 + r;
                int j = ns * 16 + ln16;
                float v = sacc[ns][r]
                        + relb[relidx[i * NN + j] * NHd + head]
                        + mask[((size_t)wr * NN + i) * NN + j];
                Sm[i * NN + j] = v;
            }
    }
    __syncthreads();

    // ---- softmax over rows (fp32), emit attn output + bf16 probs
    if (tid < NN) {
        int row = tid;
        float mx = -3.0e38f;
        for (int j = 0; j < NN; ++j) mx = fmaxf(mx, Sm[row * NN + j]);
        float sum = 0.f;
        for (int j = 0; j < NN; ++j) { float e = __expf(Sm[row * NN + j] - mx); Sm[row * NN + j] = e; sum += e; }
        float inv = 1.0f / sum;
        float* pa = attn_out + ((size_t)blk * NN + row) * NN;
        for (int j = 0; j < NN; ++j) {
            float p = Sm[row * NN + j] * inv;
            pa[j] = p;
            Pm[row * NN + j] = f2bf(p);
        }
    }
    __syncthreads();

    // ---- O = P @ V  (K=64 -> two chained WMMAs)
    {
        f32x8 oacc[2] = { zero, zero };
        #pragma unroll
        for (int k0 = 0; k0 < NN; k0 += 32) {
            U4 plo = *(const U4*)(&Pm[(wv * 16 + ln16) * NN + k0 + hlf * 8]);
            U4 phi = *(const U4*)(&Pm[(wv * 16 + ln16) * NN + k0 + 16 + hlf * 8]);
            bf16x16 ap = make_frag(plo, phi);
            #pragma unroll
            for (int ns = 0; ns < 2; ++ns) {
                U4 blo = *(const U4*)(&vT[(ns * 16 + ln16) * NN + k0 + hlf * 8]);
                U4 bhi = *(const U4*)(&vT[(ns * 16 + ln16) * NN + k0 + 16 + hlf * 8]);
                bf16x16 bv = make_frag(blo, bhi);
                oacc[ns] = __builtin_amdgcn_wmma_f32_16x16x32_bf16(
                    false, ap, false, bv, (short)0, oacc[ns], false, false);
            }
        }
        #pragma unroll
        for (int ns = 0; ns < 2; ++ns)
            #pragma unroll
            for (int r = 0; r < 8; ++r) {
                int i = wv * 16 + hlf * 8 + r;
                int c = head * HDd + ns * 16 + ln16;
                obf[(size_t)(win * NN + i) * CC + c] = f2bf(oacc[ns][r]);
            }
    }
}

// ---------------------------------------------------------------------------
// Host launcher
// ---------------------------------------------------------------------------
extern "C" void kernel_launch(void* const* d_in, const int* in_sizes, int n_in,
                              void* d_out, int out_size, void* d_ws, size_t ws_size,
                              hipStream_t stream)
{
    (void)in_sizes; (void)n_in; (void)out_size; (void)ws_size;

    const float* x      = (const float*)d_in[0];
    const float* n1g    = (const float*)d_in[1];
    const float* n1b    = (const float*)d_in[2];
    const float* qkv_w  = (const float*)d_in[3];
    const float* qkv_b  = (const float*)d_in[4];
    const float* proj_w = (const float*)d_in[5];
    const float* proj_b = (const float*)d_in[6];
    const float* rel_b  = (const float*)d_in[7];
    const float* n2g    = (const float*)d_in[8];
    const float* n2b    = (const float*)d_in[9];
    const float* fc1_w  = (const float*)d_in[10];
    const float* fc1_b  = (const float*)d_in[11];
    const float* fc2_w  = (const float*)d_in[12];
    const float* fc2_b  = (const float*)d_in[13];
    const int*   relidx = (const int*)d_in[14];
    const float* mask   = (const float*)d_in[15];

    float* y_out    = (float*)d_out;                               // [16,3136,384]
    float* attn_out = y_out + (size_t)ROWS * CC;                   // [784,12,64,64]

    // workspace carve-up (sequentially reused buffers)
    char* wp_ = (char*)d_ws;
    auto carve = [&](size_t bytes) -> char* {
        char* p = wp_; wp_ += (bytes + 255) & ~(size_t)255; return p;
    };
    u16*   buf0 = (u16*)carve((size_t)ROWS * CC * 2);     // xw -> attn-out(bf16) -> h2
    u16*   bufB = (u16*)carve((size_t)ROWS * HID * 2);    // qkv (1152 cols) -> fc1 act
    float* yb   = (float*)carve((size_t)ROWS * CC * 4);   // y after attention branch
    u16*   wq   = (u16*)carve((size_t)CC * 3 * CC * 2);   // transposed bf16 weights
    u16*   wpj  = (u16*)carve((size_t)CC * CC * 2);
    u16*   w1   = (u16*)carve((size_t)CC * HID * 2);
    u16*   w2   = (u16*)carve((size_t)HID * CC * 2);

    // weight conversion fp32 [K][N] -> bf16 transposed [N][K]
    convert_transpose_kernel<<<(CC * 3 * CC + 255) / 256, 256, 0, stream>>>(qkv_w, wq, CC, 3 * CC);
    convert_transpose_kernel<<<(CC * CC + 255) / 256, 256, 0, stream>>>(proj_w, wpj, CC, CC);
    convert_transpose_kernel<<<(CC * HID + 255) / 256, 256, 0, stream>>>(fc1_w, w1, CC, HID);
    convert_transpose_kernel<<<(HID * CC + 255) / 256, 256, 0, stream>>>(fc2_w, w2, HID, CC);

    // LN1 fused with shift + window partition -> bf16 windowed activations
    ln_kernel<<<ROWS, 128, 0, stream>>>(x, n1g, n1b, buf0, 1);

    // QKV GEMM: [50176,384] x [384,1152] -> bf16
    gemm_wmma<0><<<dim3(1152 / 128, ROWS / 128), 256, 0, stream>>>(
        buf0, wq, qkv_b, ROWS, 1152, CC, bufB, nullptr, nullptr, nullptr);

    // attention per (window, head); writes attn (fp32, d_out) and bf16 O (buf0)
    attn_kernel<<<784 * NHd, 128, 0, stream>>>(bufB, rel_b, relidx, mask, attn_out, buf0);

    // proj GEMM + window reverse + roll + residual(x) -> y (fp32)
    gemm_wmma<1><<<dim3(CC / 128, ROWS / 128), 256, 0, stream>>>(
        buf0, wpj, proj_b, ROWS, CC, CC, nullptr, yb, x, nullptr);

    // LN2 -> bf16
    ln_kernel<<<ROWS, 128, 0, stream>>>(yb, n2g, n2b, buf0, 0);

    // FC1 + GELU -> bf16
    gemm_wmma<2><<<dim3(HID / 128, ROWS / 128), 256, 0, stream>>>(
        buf0, w1, fc1_b, ROWS, HID, CC, bufB, nullptr, nullptr, nullptr);

    // FC2 + residual(y) -> fp32 final output
    gemm_wmma<3><<<dim3(CC / 128, ROWS / 128), 256, 0, stream>>>(
        bufB, w2, fc2_b, ROWS, CC, HID, nullptr, y_out, nullptr, yb);
}
